// HiPPO_61177514164456
// MI455X (gfx1250) — compile-verified, matrix operand order checked by reading, not run on hardware
//
#include <hip/hip_runtime.h>
#include <hip/hip_bf16.h>

typedef __attribute__((ext_vector_type(16))) __bf16 v16bf;
typedef __attribute__((ext_vector_type(8)))  float  v8f;

#define N64   64
#define LSEQ  524288
#define TBLK  64
#define MB    (LSEQ / TBLK)   /* 8192 blocks  */
#define KGRP  64              /* blocks per group */
#define NGRP  (MB / KGRP)     /* 128 groups   */
#define MAT   (N64 * N64)     /* 4096 elems per 64x64 matrix */
#define CSTRIDE 4100          /* padded per-block stride in Cbuf (4100 % 64 == 4) */

// Swizzled fragment layout: a 16x32 (A) or 32x16 (B) bf16 tile occupies 512
// contiguous bf16 = 32 lanes x 16 halves, each lane's 16 values contiguous
// (32 B). Mapping for element with row-in-tile r (A: M, B: N) and K index k:
//   lane = (r & 15) + ((k & 8) ? 16 : 0)
//   h    = (k & 7) + (((k >> 4) & 1) << 3)
__device__ __forceinline__ int frag_slot(int r, int k) {
  int lane = (r & 15) + ((k & 8) ? 16 : 0);
  int h = (k & 7) + (((k >> 4) & 1) << 3);
  return lane * 16 + h;
}

// ---------------- Stage 1: matrix powers (f32, precision critical) ----------
__global__ void k_init_powers(const float* __restrict__ Ab, float* __restrict__ P) {
  int tid = blockIdx.x * blockDim.x + threadIdx.x;
  if (tid >= MAT) return;
  int n = tid >> 6, k = tid & 63;
  P[tid]       = (n == k) ? 1.0f : 0.0f;  // P[0] = I
  P[MAT + tid] = Ab[tid];                 // P[1] = Ab
}

__global__ void k_q_init(const float* __restrict__ P, float* __restrict__ Q) {
  int tid = blockIdx.x * blockDim.x + threadIdx.x;
  if (tid >= MAT) return;
  int n = tid >> 6, k = tid & 63;
  Q[tid]       = (n == k) ? 1.0f : 0.0f;       // Q[0] = I
  Q[MAT + tid] = P[(size_t)TBLK * MAT + tid];  // Q[1] = Ab^64
}

// Doubling stage: tab[half + i] = tab[half] @ tab[i], i = 1..half (powers commute)
__global__ void k_pow_stage(float* __restrict__ tab, int half) {
  __shared__ float As[MAT];
  __shared__ float Bs[MAT];
  int i = blockIdx.x + 1;
  const float* A = tab + (size_t)half * MAT;
  const float* B = tab + (size_t)i * MAT;
  float*       C = tab + (size_t)(half + i) * MAT;
  int tid = threadIdx.x;
#pragma unroll
  for (int e = 0; e < 16; ++e) { As[tid * 16 + e] = A[tid * 16 + e]; Bs[tid * 16 + e] = B[tid * 16 + e]; }
  __syncthreads();
  int n  = tid >> 2;
  int kc = (tid & 3) * 16;
  float acc[16];
#pragma unroll
  for (int e = 0; e < 16; ++e) acc[e] = 0.0f;
  for (int r = 0; r < 64; ++r) {
    float a = As[n * 64 + r];
#pragma unroll
    for (int e = 0; e < 16; ++e) acc[e] += a * Bs[r * 64 + kc + e];
  }
#pragma unroll
  for (int e = 0; e < 16; ++e) C[n * 64 + kc + e] = acc[e];
}

// W[j] = P_j b : f32 linear + bf16 swizzled (B-operand tiles: k=j, col=n)
__global__ void k_compute_W(const float* __restrict__ P, const float* __restrict__ Bb,
                            float* __restrict__ W, __bf16* __restrict__ Wsw) {
  int tid = blockIdx.x * blockDim.x + threadIdx.x;
  if (tid >= TBLK * N64) return;
  int j = tid >> 6, n = tid & 63;
  float acc = 0.0f;
  for (int k = 0; k < 64; ++k) acc += P[(size_t)j * MAT + n * 64 + k] * Bb[k];
  W[j * 64 + n] = acc;
  int ntile = n >> 4, k0 = j >> 5;
  Wsw[(size_t)(ntile * 2 + k0) * 512 + frag_slot(n, j & 31)] = (__bf16)acc;
}

// P -> bf16, swizzled as A-operand tiles (row=n, k)
__global__ void k_swizzle_P(const float* __restrict__ P, __bf16* __restrict__ Psw, int total) {
  int tid = blockIdx.x * blockDim.x + threadIdx.x;
  if (tid >= total) return;
  int t = tid >> 12, r = tid & 4095;
  int n = r >> 6, k = r & 63;
  int ntile = n >> 4, k0 = k >> 5;
  Psw[(size_t)t * MAT + (size_t)(ntile * 2 + k0) * 512 + frag_slot(n, k & 31)] = (__bf16)P[tid];
}

// ---------------- Stage 2: block / group / top scans (f32) ------------------
// c_m[n] = sum_j W[63-j][n] * u[m*64+j]  -> C2[n*MB + m]
__global__ void k_compute_C2(const float* __restrict__ u, const float* __restrict__ W,
                             float* __restrict__ C2) {
  __shared__ float ub[256];
  int tid = threadIdx.x;
  ub[tid] = u[(size_t)blockIdx.x * 256 + tid];
  __syncthreads();
  int mi = tid >> 6, n = tid & 63;
  int m = blockIdx.x * 4 + mi;
  float acc = 0.0f;
  for (int j = 0; j < 64; ++j) acc += W[(63 - j) * 64 + n] * ub[mi * 64 + j];
  C2[(size_t)n * MB + m] = acc;
}

// D[:,g] = sum_i Q[63-i] @ C2[:, g*64+i]
__global__ void k_compute_Dg(const float* __restrict__ Q, const float* __restrict__ C2,
                             float* __restrict__ Dg) {
  __shared__ float Cg[MAT];
  int g = blockIdx.x, n = threadIdx.x;
  for (int i = 0; i < 64; ++i) Cg[n * 64 + i] = C2[(size_t)n * MB + g * 64 + i];
  __syncthreads();
  float acc = 0.0f;
  for (int i = 0; i < 64; ++i) {
    const float* Qr = Q + (size_t)(63 - i) * MAT + n * 64;
    for (int k = 0; k < 64; ++k) acc += Qr[k] * Cg[k * 64 + i];
  }
  Dg[n * NGRP + g] = acc;
}

// Sequential top scan over 128 groups: sg' = Q[64] sg + D[:,g]
__global__ void k_top_scan(const float* __restrict__ Q, const float* __restrict__ Dg,
                           const float* __restrict__ x0, float* __restrict__ SG) {
  __shared__ float QK[MAT];
  __shared__ float s[64];
  int n = threadIdx.x;
  for (int e = 0; e < 64; ++e) QK[n * 64 + e] = Q[(size_t)KGRP * MAT + n * 64 + e];
  s[n] = x0[n];
  __syncthreads();
  for (int g = 0; g < NGRP; ++g) {
    SG[n * NGRP + g] = s[n];
    float acc = Dg[n * NGRP + g];
    for (int k = 0; k < 64; ++k) acc += QK[n * 64 + k] * s[k];
    __syncthreads();
    s[n] = acc;
    __syncthreads();
  }
}

// Per-group fine scan: 64 steps of s' = Ab^64 s + c_m ; emit bf16 states
// directly in swizzled B-operand layout (k = state dim n, col = block m).
__global__ void k_fine_states(const float* __restrict__ Q, const float* __restrict__ C2,
                              const float* __restrict__ SG, __bf16* __restrict__ Ssw) {
  __shared__ float P64s[MAT];
  __shared__ float s[64];
  int g = blockIdx.x, n = threadIdx.x;
  for (int e = 0; e < 64; ++e) P64s[n * 64 + e] = Q[(size_t)MAT + n * 64 + e];  // Q[1]=Ab^64
  s[n] = SG[n * NGRP + g];
  __syncthreads();
  int k0 = n >> 5, klocal = n & 31;
  for (int i = 0; i < KGRP; ++i) {
    int m = g * KGRP + i;
    int mt = m >> 4;
    Ssw[(size_t)(mt * 2 + k0) * 512 + frag_slot(m, klocal)] = (__bf16)s[n];
    float acc = C2[(size_t)n * MB + m];
    for (int k = 0; k < 64; ++k) acc += P64s[n * 64 + k] * s[k];
    __syncthreads();
    s[n] = acc;
    __syncthreads();
  }
}

// ---------------- Stage 3: fused y kernel (bf16 WMMA, single y write) -------
// Workgroup owns 16 consecutive blocks (one 32x16 B-fragment pair of states).
// Phase 1: conv part per block -> f32 LDS buffer Cbuf (256 KB, CDNA5 big LDS).
// Phase 2: per-wave-owned t values: P^{t+1} @ States, add Cbuf, store y once.
__global__ void k_y_fused(const float* __restrict__ u, const __bf16* __restrict__ Psw,
                          const __bf16* __restrict__ Wsw, const __bf16* __restrict__ Ssw,
                          float* __restrict__ y) {
  __shared__ float  Cbuf[16 * CSTRIDE];   // 262400 B
  __shared__ __bf16 Mu[2 * MAT];          //  16384 B (double-buffered Toeplitz)
  __shared__ __bf16 urev[16 * 128];       //   4096 B (reversed zero-padded bf16 u)
  int mtile = blockIdx.x;                 // 16 blocks: m = mtile*16 .. +15
  int tid = threadIdx.x;
  int wave = tid >> 5, lane = tid & 31;
  int l16 = lane & 15, hi = (lane & 16) ? 8 : 0;

  // stage reversed zero-padded bf16 u: urev[b*128 + o] = (o<=63) ? u_b[63-o] : 0
  {
    const float* ublk = u + (size_t)mtile * 1024;
#pragma unroll
    for (int e = 0; e < 8; ++e) {
      int i = tid * 8 + e;
      int o = i & 127, b = i >> 7;
      urev[i] = (o <= 63) ? (__bf16)ublk[b * 64 + 63 - o] : (__bf16)0.0f;
    }
  }

  // this workgroup's two state B-fragments (resident all kernel)
  const v16bf* Bsrc = (const v16bf*)Ssw;
  v16bf sb0 = Bsrc[(size_t)(mtile * 2 + 0) * 32 + lane];
  v16bf sb1 = Bsrc[(size_t)(mtile * 2 + 1) * 32 + lane];

  // this wave's W B-fragments for phase 1 (2 n-tiles x 2 k-steps)
  const v16bf* Wf = (const v16bf*)Wsw;
  int nn0 = (wave * 2) & 3, nn1 = (wave * 2 + 1) & 3;
  int tt = (wave * 2) >> 2;               // same for both of this wave's tiles
  v16bf wb00 = Wf[(nn0 * 2 + 0) * 32 + lane];
  v16bf wb01 = Wf[(nn0 * 2 + 1) * 32 + lane];
  v16bf wb10 = Wf[(nn1 * 2 + 0) * 32 + lane];
  v16bf wb11 = Wf[(nn1 * 2 + 1) * 32 + lane];

  __builtin_prefetch(Psw + (size_t)(wave + 1) * MAT + lane * 16, 0, 1);

  // per-thread invariants of the Mu fill (thread owns chunk Mu[tid*16..+15]);
  // ascending gather start in the reversed array:
  int f_lane = tid & 31, f_tl = tid >> 5;
  int f_trow = (f_tl >> 1) * 16 + (f_lane & 15);
  int f_base = (f_tl & 1) * 32 + ((f_lane & 16) ? 8 : 0);
  int r0 = 63 - f_trow + f_base;          // run: urev[r0..r0+7], urev[r0+16..r0+23]

  __syncthreads();

  // ---- Phase 1: conv per block into Cbuf (double-buffered Mu, 1 barrier) ---
  for (int b = 0; b < 16; ++b) {
    __bf16* mbuf = Mu + (b & 1) * MAT;
    {
      const __bf16* up = urev + b * 128 + r0;
      v16bf mv;
#pragma unroll
      for (int e = 0; e < 8; ++e) mv[e] = up[e];
#pragma unroll
      for (int e = 0; e < 8; ++e) mv[8 + e] = up[16 + e];
      *(v16bf*)(mbuf + tid * 16) = mv;
    }
    __syncthreads();
    v8f acc0 = {}, acc1 = {};
    {
      v16bf a0 = *(const v16bf*)(mbuf + ((tt * 2 + 0) * 32 + lane) * 16);
      v16bf a1 = *(const v16bf*)(mbuf + ((tt * 2 + 1) * 32 + lane) * 16);
      acc0 = __builtin_amdgcn_wmma_f32_16x16x32_bf16(false, a0, false, wb00, (short)0, acc0, false, false);
      acc1 = __builtin_amdgcn_wmma_f32_16x16x32_bf16(false, a0, false, wb10, (short)0, acc1, false, false);
      acc0 = __builtin_amdgcn_wmma_f32_16x16x32_bf16(false, a1, false, wb01, (short)0, acc0, false, false);
      acc1 = __builtin_amdgcn_wmma_f32_16x16x32_bf16(false, a1, false, wb11, (short)0, acc1, false, false);
    }
    float* cdst = Cbuf + b * CSTRIDE;
#pragma unroll
    for (int r = 0; r < 8; ++r) {
      int trow = tt * 16 + hi + r;
      cdst[trow * 64 + nn0 * 16 + l16] = acc0[r];
      cdst[trow * 64 + nn1 * 16 + l16] = acc1[r];
    }
  }
  __syncthreads();

  // ---- Phase 2: state part + combine + single store of y ----
#pragma unroll 1
  for (int it = 0; it < 8; ++it) {
    int t = wave + 8 * it;                       // each wave owns 8 t values
    const v16bf* Asrc = (const v16bf*)(Psw + (size_t)(t + 1) * MAT);
#pragma unroll
    for (int q = 0; q < 4; ++q) {                // n tiles
      v8f acc = {};
      v16bf a0 = Asrc[(q * 2 + 0) * 32 + lane];
      acc = __builtin_amdgcn_wmma_f32_16x16x32_bf16(false, a0, false, sb0, (short)0, acc, false, false);
      v16bf a1 = Asrc[(q * 2 + 1) * 32 + lane];
      acc = __builtin_amdgcn_wmma_f32_16x16x32_bf16(false, a1, false, sb1, (short)0, acc, false, false);
      int nb = q * 16 + hi;
      const float* csrc = Cbuf + l16 * CSTRIDE + t * 64 + nb;  // lane-contiguous 32 B
      int mglob = mtile * 16 + l16;
      size_t idx = ((size_t)mglob * TBLK + t) * N64 + nb;
      float4 lo = make_float4(acc[0] + csrc[0], acc[1] + csrc[1],
                              acc[2] + csrc[2], acc[3] + csrc[3]);
      float4 hi4 = make_float4(acc[4] + csrc[4], acc[5] + csrc[5],
                               acc[6] + csrc[6], acc[7] + csrc[7]);
      *(float4*)(y + idx)     = lo;
      *(float4*)(y + idx + 4) = hi4;
    }
  }
}

// ---------------- host-side orchestration -----------------------------------
extern "C" void kernel_launch(void* const* d_in, const int* in_sizes, int n_in,
                              void* d_out, int out_size, void* d_ws, size_t ws_size,
                              hipStream_t stream) {
  const float* u  = (const float*)d_in[0];
  const float* Ab = (const float*)d_in[1];
  const float* Bb = (const float*)d_in[2];
  const float* x0 = (const float*)d_in[3];
  float* y = (float*)d_out;

  // workspace carve-up
  float* P   = (float*)d_ws;                 // 65 * 4096 f32
  float* Q   = P + 65 * MAT;                 // 65 * 4096 f32
  float* W   = Q + 65 * MAT;                 // 4096 f32
  float* C2  = W + MAT;                      // 64*8192 f32
  float* Dg  = C2 + (size_t)N64 * MB;        // 64*128 f32
  float* SG  = Dg + N64 * NGRP;              // 64*128 f32
  __bf16* Psw = (__bf16*)(SG + N64 * NGRP);  // 65*4096 bf16 (swizzled A tiles)
  __bf16* Wsw = Psw + 65 * MAT;              // 4096 bf16    (swizzled B tiles)
  __bf16* Ssw = Wsw + MAT;                   // 64*8192 bf16 (swizzled B tiles)

  // 1) powers of Ab (fine) and Ab^64 (coarse) by log-doubling
  k_init_powers<<<16, 256, 0, stream>>>(Ab, P);
  for (int s = 0; s < 6; ++s)
    k_pow_stage<<<(1 << s), 256, 0, stream>>>(P, 1 << s);
  k_q_init<<<16, 256, 0, stream>>>(P, Q);
  for (int s = 0; s < 6; ++s)
    k_pow_stage<<<(1 << s), 256, 0, stream>>>(Q, 1 << s);

  // 2) impulse-response rows W and swizzled bf16 tables
  k_compute_W<<<16, 256, 0, stream>>>(P, Bb, W, Wsw);
  k_swizzle_P<<<(65 * MAT + 255) / 256, 256, 0, stream>>>(P, Psw, 65 * MAT);

  // 3) block contributions, group contributions, hierarchical scans
  k_compute_C2<<<MB / 4, 256, 0, stream>>>(u, W, C2);
  k_compute_Dg<<<NGRP, 64, 0, stream>>>(Q, C2, Dg);
  k_top_scan<<<1, 64, 0, stream>>>(Q, Dg, x0, SG);
  k_fine_states<<<NGRP, 64, 0, stream>>>(Q, C2, SG, Ssw);

  // 4) fused WMMA pass: y written exactly once (~131 MB HBM total)
  k_y_fused<<<MB / 16, 256, 0, stream>>>(u, Psw, Wsw, Ssw, y);

  (void)in_sizes; (void)n_in; (void)out_size; (void)ws_size;
}
